// BiMambaWrapper_773094113548
// MI455X (gfx1250) — matrix operand order repaired and checked by reading, not compiled, not run on hardware
//
#include <hip/hip_runtime.h>
#include <hip/hip_bf16.h>

// ---------------- problem constants ----------------
#define D_MODEL 512
#define D_INNER 1024
#define D_STATE 16
#define DT_RANK 32
#define B_SZ 2
#define SEQ 8192
#define NTOK (B_SZ * SEQ)          // 16384 tokens
#define XPROJ_OUT (DT_RANK + 2 * D_STATE)  // 64
#define CHUNK 128
#define NCHUNK (SEQ / CHUNK)       // 64

typedef __bf16 bf16;
typedef __attribute__((ext_vector_type(16))) __bf16 v16bf;
typedef __attribute__((ext_vector_type(8)))  __bf16 v8bf;
typedef __attribute__((ext_vector_type(8)))  float  v8f;

union V16U { v16bf v; v8bf h[2]; };

// ---------------- CDNA5 async global->LDS copy (ASYNCcnt path) ----------------
// 16B per lane, per-lane LDS destination address in a VGPR (addr[31:0] of the
// generic shared pointer is the LDS byte offset).
__device__ __forceinline__ void async_copy16(void* lds_ptr, const void* gptr) {
    unsigned lds_addr = (unsigned)(size_t)lds_ptr;
    asm volatile("global_load_async_to_lds_b128 %0, %1, off"
                 :: "v"(lds_addr), "v"(gptr) : "memory");
}
__device__ __forceinline__ void wait_async0() {
    asm volatile("s_wait_asynccnt 0x0" ::: "memory");
}

// ---------------- fp32 -> bf16 convert ----------------
__global__ void cvt_f32_bf16(const float* __restrict__ src, bf16* __restrict__ dst, int n) {
    int i = blockIdx.x * blockDim.x + threadIdx.x;
    if (i < n) dst[i] = (bf16)src[i];
}

// ---------------- WMMA GEMM: C[M,N] = A[M,K] * W[N,K]^T ----------------
// A row-major bf16 (optionally summed with A2 while staging), W row-major bf16
// (out_features x in_features), C fp32.
// Block tile 128x64, 8 waves as 4(M) x 2(N); wave tile 32x32 = 4 WMMA / K-step.
// Double-buffered LDS, one barrier per K-step. Non-SUMA path uses
// GLOBAL_LOAD_ASYNC_TO_LDS_B128 (direct global->LDS, ASYNCcnt tracked).
#define BM 128
#define BN 64
#define BK 32
template<bool SUMA>
__global__ __launch_bounds__(256) void wmma_gemm_bf16(
    const bf16* __restrict__ A, const bf16* __restrict__ A2,
    const bf16* __restrict__ W, float* __restrict__ C,
    int M, int N, int K)
{
    __shared__ bf16 lA[2][BM][BK];   // 16 KB
    __shared__ bf16 lB[2][BN][BK];   //  8 KB
    const int bm = blockIdx.x * BM;
    const int bn = blockIdx.y * BN;
    const int tid  = threadIdx.x;
    const int wave = tid >> 5;
    const int lane = tid & 31;
    const int wm = (wave & 3) * 32;          // 0,32,64,96
    const int wn = (wave >> 2) * 32;         // 0,32
    const int arow  = lane & 15;
    const int hiHalf = (lane >> 4) & 1;      // lanes 16-31
    const int ka = hiHalf * 8;               // A frag K base: 0 or 8
    const int kb = hiHalf * 8;               // B frag K base: 0 or 8

    v8f acc00 = {}, acc10 = {}, acc01 = {}, acc11 = {};

    // global->LDS staging map: each thread moves 2x16B of A and 1x16B of B
    const int ldr = tid >> 2;                // 0..63
    const int ldc = (tid & 3) * 8;           // 0,8,16,24

    const bf16* gA0 = A + (size_t)(bm + ldr) * K + ldc;
    const bf16* gA1 = A + (size_t)(bm + ldr + 64) * K + ldc;
    const bf16* gB  = W + (size_t)(bn + ldr) * K + ldc;

    auto compute = [&](int buf) {
        V16U a0, a1, b0, b1;
        a0.h[0] = *(const v8bf*)&lA[buf][wm + arow][ka];
        a0.h[1] = *(const v8bf*)&lA[buf][wm + arow][ka + 16];
        a1.h[0] = *(const v8bf*)&lA[buf][wm + 16 + arow][ka];
        a1.h[1] = *(const v8bf*)&lA[buf][wm + 16 + arow][ka + 16];
        b0.h[0] = *(const v8bf*)&lB[buf][wn + arow][kb];
        b0.h[1] = *(const v8bf*)&lB[buf][wn + arow][kb + 16];
        b1.h[0] = *(const v8bf*)&lB[buf][wn + 16 + arow][kb];
        b1.h[1] = *(const v8bf*)&lB[buf][wn + 16 + arow][kb + 16];
        acc00 = __builtin_amdgcn_wmma_f32_16x16x32_bf16(false, a0.v, false, b0.v,
                                                        (short)0, acc00, false, false);
        acc10 = __builtin_amdgcn_wmma_f32_16x16x32_bf16(false, a1.v, false, b0.v,
                                                        (short)0, acc10, false, false);
        acc01 = __builtin_amdgcn_wmma_f32_16x16x32_bf16(false, a0.v, false, b1.v,
                                                        (short)0, acc01, false, false);
        acc11 = __builtin_amdgcn_wmma_f32_16x16x32_bf16(false, a1.v, false, b1.v,
                                                        (short)0, acc11, false, false);
    };

    int buf = 0;
    if constexpr (!SUMA) {
        // -------- async-copy pipeline --------
        async_copy16(&lA[0][ldr][ldc],      gA0);
        async_copy16(&lA[0][ldr + 64][ldc], gA1);
        async_copy16(&lB[0][ldr][ldc],      gB);
        wait_async0();
        __syncthreads();
        for (int k0 = BK; k0 < K; k0 += BK) {
            // fill the other buffer via the async engine while WMMAs run
            async_copy16(&lA[buf ^ 1][ldr][ldc],      gA0 + k0);
            async_copy16(&lA[buf ^ 1][ldr + 64][ldc], gA1 + k0);
            async_copy16(&lB[buf ^ 1][ldr][ldc],      gB + k0);
            compute(buf);
            wait_async0();
            __syncthreads();
            buf ^= 1;
        }
        compute(buf);
    } else {
        // -------- register-staged pipeline (A = A1 + A2 fused while staging) --------
        auto loadA = [&](const bf16* g, int k0) -> v8bf {
            v8bf x = *(const v8bf*)(g + k0);
            v8bf y = *(const v8bf*)(A2 + (size_t)(g - A) + k0);
#pragma unroll
            for (int i = 0; i < 8; ++i) x[i] = (bf16)((float)x[i] + (float)y[i]);
            return x;
        };
        v8bf sa0 = loadA(gA0, 0);
        v8bf sa1 = loadA(gA1, 0);
        v8bf sb  = *(const v8bf*)gB;
        *(v8bf*)&lA[0][ldr][ldc]      = sa0;
        *(v8bf*)&lA[0][ldr + 64][ldc] = sa1;
        *(v8bf*)&lB[0][ldr][ldc]      = sb;
        __syncthreads();
        for (int k0 = BK; k0 < K; k0 += BK) {
            sa0 = loadA(gA0, k0);
            sa1 = loadA(gA1, k0);
            sb  = *(const v8bf*)(gB + k0);
            compute(buf);
            *(v8bf*)&lA[buf ^ 1][ldr][ldc]      = sa0;
            *(v8bf*)&lA[buf ^ 1][ldr + 64][ldc] = sa1;
            *(v8bf*)&lB[buf ^ 1][ldr][ldc]      = sb;
            __syncthreads();
            buf ^= 1;
        }
        compute(buf);
    }

    // C/D layout: n = lane&15 ; m = vgpr + 8*(lane>=16)
    const int cn = bn + wn + arow;
    const int cm = bm + wm + hiHalf * 8;
#pragma unroll
    for (int r = 0; r < 8; ++r) {
        C[(size_t)(cm + r) * N + cn]           = acc00[r];
        C[(size_t)(cm + 16 + r) * N + cn]      = acc10[r];
        C[(size_t)(cm + r) * N + cn + 16]      = acc01[r];
        C[(size_t)(cm + 16 + r) * N + cn + 16] = acc11[r];
    }
}

// ---------------- depthwise conv (causal fwd, anti-causal rev) + SiLU ----------------
__device__ __forceinline__ float silu_f(float x) { return x / (1.f + __expf(-x)); }

__global__ void conv_silu_kernel(const float* __restrict__ xz,
                                 const float* __restrict__ cwf, const float* __restrict__ cbf,
                                 const float* __restrict__ cwr, const float* __restrict__ cbr,
                                 bf16* __restrict__ xf, bf16* __restrict__ xr)
{
    int idx = blockIdx.x * blockDim.x + threadIdx.x;   // over B*SEQ*D_INNER
    int e = idx & (D_INNER - 1);
    int t = (idx >> 10) & (SEQ - 1);
    int b = idx >> 23;
    float accf = cbf[e], accr = cbr[e];
#pragma unroll
    for (int k = 0; k < 4; ++k) {
        int tf = t - 3 + k;
        if (tf >= 0)
            accf += cwf[e * 4 + k] * xz[((size_t)b * SEQ + tf) * (2 * D_INNER) + e];
        int tr = t + 3 - k;
        if (tr < SEQ)
            accr += cwr[e * 4 + k] * xz[((size_t)b * SEQ + tr) * (2 * D_INNER) + e];
    }
    xf[idx] = (bf16)silu_f(accf);
    xr[idx] = (bf16)silu_f(accr);
}

// ---------------- chunked selective scan ----------------
// PASS 1: per-chunk (prod dA, local final h). PASS 2: replay with carry-in, emit gated y.
template<int PASS>
__global__ __launch_bounds__(256) void scan_kernel(
    const float* __restrict__ xz,
    const bf16* __restrict__ xf, const bf16* __restrict__ xr,
    const float* __restrict__ xdf, const float* __restrict__ xdr,
    const float* __restrict__ dtw_f, const float* __restrict__ dtb_f,
    const float* __restrict__ Alog_f, const float* __restrict__ Dp_f,
    const float* __restrict__ dtw_r, const float* __restrict__ dtb_r,
    const float* __restrict__ Alog_r, const float* __restrict__ Dp_r,
    float* __restrict__ chA, float* __restrict__ chH,
    bf16* __restrict__ yf, bf16* __restrict__ yr)
{
    const int tid   = threadIdx.x;
    const int e     = blockIdx.x * 256 + tid;
    const int chunk = blockIdx.y;
    const int dirb  = blockIdx.z;            // dir*2 + b
    const int dir   = dirb >> 1;
    const int b     = dirb & 1;

    const float* dtw  = dir ? dtw_r  : dtw_f;
    const float* dtbp = dir ? dtb_r  : dtb_f;
    const float* Alog = dir ? Alog_r : Alog_f;
    const float* Dp   = dir ? Dp_r   : Dp_f;
    const bf16*  xg   = dir ? xr : xf;
    const float* xdbl = dir ? xdr : xdf;
    bf16*        yout = dir ? yr : yf;

    float w[DT_RANK];
#pragma unroll
    for (int r = 0; r < DT_RANK; ++r) w[r] = dtw[e * DT_RANK + r];
    float Aa[D_STATE];
#pragma unroll
    for (int n = 0; n < D_STATE; ++n) Aa[n] = -__expf(Alog[e * D_STATE + n]);
    const float dtb = dtbp[e];
    const float Dd  = Dp[e];

    const size_t cbase = (((size_t)dirb * D_INNER + e) * NCHUNK + chunk) * D_STATE;
    float h[D_STATE], P[D_STATE];
#pragma unroll
    for (int n = 0; n < D_STATE; ++n) {
        if (PASS == 1) { h[n] = 0.f; P[n] = 1.f; }
        else           { h[n] = chH[cbase + n]; P[n] = 0.f; }
    }

    __shared__ float sxd[16][XPROJ_OUT];     // 16 timesteps of xdbl rows (4 KB)
    const int srow = tid >> 4;               // staging row 0..15
    const int scol = (tid & 15) * 4;         // staging 4 floats

    for (int sb = 0; sb < CHUNK / 16; ++sb) {
        // stage 16 rows of xdbl (dt|B|C) for this sub-block
        {
            int i0 = chunk * CHUNK + sb * 16 + srow;
            int tg = (dir == 0) ? i0 : (SEQ - 1 - i0);
            *(float4*)&sxd[srow][scol] =
                *(const float4*)&xdbl[((size_t)b * SEQ + tg) * XPROJ_OUT + scol];
        }
        __syncthreads();
#pragma unroll 1
        for (int i = 0; i < 16; ++i) {
            int i0 = chunk * CHUNK + sb * 16 + i;
            int t  = (dir == 0) ? i0 : (SEQ - 1 - i0);
            float dlin = dtb;
#pragma unroll
            for (int r = 0; r < DT_RANK; ++r) dlin += sxd[i][r] * w[r];
            float delta = (dlin > 20.f) ? dlin : logf(1.f + __expf(dlin));
            float xv  = (float)xg[((size_t)b * SEQ + t) * D_INNER + e];
            float dbx = delta * xv;
            float acc = 0.f;
#pragma unroll
            for (int n = 0; n < D_STATE; ++n) {
                float da = __expf(delta * Aa[n]);
                h[n] = da * h[n] + dbx * sxd[i][DT_RANK + n];
                if (PASS == 1) P[n] *= da;
                else           acc += h[n] * sxd[i][DT_RANK + D_STATE + n];
            }
            if (PASS == 2) {
                float zv = xz[((size_t)b * SEQ + t) * (2 * D_INNER) + D_INNER + e];
                float out = (acc + xv * Dd) * silu_f(zv);
                yout[((size_t)b * SEQ + t) * D_INNER + e] = (bf16)out;
            }
        }
        __syncthreads();
    }
    if (PASS == 1) {
#pragma unroll
        for (int n = 0; n < D_STATE; ++n) { chA[cbase + n] = P[n]; chH[cbase + n] = h[n]; }
    }
}

// ---------------- middle scan over chunk states ----------------
// chH[...] is rewritten in-place to the EXCLUSIVE carry-in for each chunk.
__global__ void scan_mid_kernel(const float* __restrict__ chA, float* __restrict__ chH)
{
    int idx = blockIdx.x * blockDim.x + threadIdx.x;   // 4 * D_INNER * D_STATE
    int n    = idx & (D_STATE - 1);
    int e    = (idx >> 4) & (D_INNER - 1);
    int dirb = idx >> 14;
    size_t base = ((size_t)dirb * D_INNER + e) * NCHUNK * D_STATE + n;
    float H = 0.f;
    for (int c = 0; c < NCHUNK; ++c) {
        size_t o = base + (size_t)c * D_STATE;
        float Pv = chA[o];
        float hl = chH[o];
        chH[o] = H;                    // carry-in for chunk c
        H = Pv * H + hl;
    }
}

// ---------------- launcher ----------------
static inline size_t alignup(size_t x) { return (x + 255) & ~(size_t)255; }

extern "C" void kernel_launch(void* const* d_in, const int* in_sizes, int n_in,
                              void* d_out, int out_size, void* d_ws, size_t ws_size,
                              hipStream_t stream) {
    const float* u     = (const float*)d_in[0];
    const float* Win   = (const float*)d_in[1];
    const float* Wout  = (const float*)d_in[2];
    const float* cwf   = (const float*)d_in[3];
    const float* cbf   = (const float*)d_in[4];
    const float* xpf   = (const float*)d_in[5];
    const float* dtwf  = (const float*)d_in[6];
    const float* dtbf  = (const float*)d_in[7];
    const float* Alf   = (const float*)d_in[8];
    const float* Dfp   = (const float*)d_in[9];
    const float* cwr   = (const float*)d_in[10];
    const float* cbr   = (const float*)d_in[11];
    const float* xpr   = (const float*)d_in[12];
    const float* dtwr  = (const float*)d_in[13];
    const float* dtbr  = (const float*)d_in[14];
    const float* Alr   = (const float*)d_in[15];
    const float* Drp   = (const float*)d_in[16];

    char* ws = (char*)d_ws;
    size_t off = 0;
    float* xz      = (float*)(ws + off); off = alignup(off + (size_t)NTOK * 2 * D_INNER * 4);
    bf16*  u_bf    = (bf16*)(ws + off);  off = alignup(off + (size_t)NTOK * D_MODEL * 2);
    bf16*  Win_bf  = (bf16*)(ws + off);  off = alignup(off + (size_t)2 * D_INNER * D_MODEL * 2);
    bf16*  Wout_bf = (bf16*)(ws + off);  off = alignup(off + (size_t)D_MODEL * D_INNER * 2);
    bf16*  xpF_bf  = (bf16*)(ws + off);  off = alignup(off + (size_t)XPROJ_OUT * D_INNER * 2);
    bf16*  xpR_bf  = (bf16*)(ws + off);  off = alignup(off + (size_t)XPROJ_OUT * D_INNER * 2);
    bf16*  xf_bf   = (bf16*)(ws + off);  off = alignup(off + (size_t)NTOK * D_INNER * 2);
    bf16*  xr_bf   = (bf16*)(ws + off);  off = alignup(off + (size_t)NTOK * D_INNER * 2);
    float* xdbl_f  = (float*)(ws + off); off = alignup(off + (size_t)NTOK * XPROJ_OUT * 4);
    float* xdbl_r  = (float*)(ws + off); off = alignup(off + (size_t)NTOK * XPROJ_OUT * 4);
    float* chA     = (float*)(ws + off); off = alignup(off + (size_t)4 * D_INNER * NCHUNK * D_STATE * 4);
    float* chH     = (float*)(ws + off); off = alignup(off + (size_t)4 * D_INNER * NCHUNK * D_STATE * 4);
    bf16*  y_f     = (bf16*)(ws + off);  off = alignup(off + (size_t)NTOK * D_INNER * 2);
    bf16*  y_r     = (bf16*)(ws + off);  off = alignup(off + (size_t)NTOK * D_INNER * 2);

    // 1) bf16 conversions (activations + weights)
    {
        int n = NTOK * D_MODEL;
        cvt_f32_bf16<<<(n + 255) / 256, 256, 0, stream>>>(u, u_bf, n);
        n = 2 * D_INNER * D_MODEL;
        cvt_f32_bf16<<<(n + 255) / 256, 256, 0, stream>>>(Win, Win_bf, n);
        n = D_MODEL * D_INNER;
        cvt_f32_bf16<<<(n + 255) / 256, 256, 0, stream>>>(Wout, Wout_bf, n);
        n = XPROJ_OUT * D_INNER;
        cvt_f32_bf16<<<(n + 255) / 256, 256, 0, stream>>>(xpf, xpF_bf, n);
        cvt_f32_bf16<<<(n + 255) / 256, 256, 0, stream>>>(xpr, xpR_bf, n);
    }

    // 2) in_proj (shared by both directions): xz = u @ Win^T   [16384 x 2048]
    wmma_gemm_bf16<false><<<dim3(NTOK / BM, (2 * D_INNER) / BN), 256, 0, stream>>>(
        u_bf, nullptr, Win_bf, xz, NTOK, 2 * D_INNER, D_MODEL);

    // 3) depthwise conv (fwd causal / rev anti-causal) + SiLU -> bf16
    conv_silu_kernel<<<(NTOK * D_INNER) / 256, 256, 0, stream>>>(
        xz, cwf, cbf, cwr, cbr, xf_bf, xr_bf);

    // 4) x_proj for both directions: xdbl = x @ xproj^T   [16384 x 64]
    wmma_gemm_bf16<false><<<dim3(NTOK / BM, XPROJ_OUT / BN), 256, 0, stream>>>(
        xf_bf, nullptr, xpF_bf, xdbl_f, NTOK, XPROJ_OUT, D_INNER);
    wmma_gemm_bf16<false><<<dim3(NTOK / BM, XPROJ_OUT / BN), 256, 0, stream>>>(
        xr_bf, nullptr, xpR_bf, xdbl_r, NTOK, XPROJ_OUT, D_INNER);

    // 5) chunked selective scan (both directions in one launch)
    dim3 gs(D_INNER / 256, NCHUNK, 4);   // egroup x chunk x (dir*2+b)
    scan_kernel<1><<<gs, 256, 0, stream>>>(
        xz, xf_bf, xr_bf, xdbl_f, xdbl_r,
        dtwf, dtbf, Alf, Dfp, dtwr, dtbr, Alr, Drp,
        chA, chH, y_f, y_r);
    scan_mid_kernel<<<(4 * D_INNER * D_STATE) / 256, 256, 0, stream>>>(chA, chH);
    scan_kernel<2><<<gs, 256, 0, stream>>>(
        xz, xf_bf, xr_bf, xdbl_f, xdbl_r,
        dtwf, dtbf, Alf, Dfp, dtwr, dtbr, Alr, Drp,
        chA, chH, y_f, y_r);

    // 6) out_proj with fused (y_f + y_r): out = (y_f + y_r) @ Wout^T  [16384 x 512] fp32
    wmma_gemm_bf16<true><<<dim3(NTOK / BM, D_MODEL / BN), 256, 0, stream>>>(
        y_f, y_r, Wout_bf, (float*)d_out, NTOK, D_MODEL, D_INNER);
}
